// SatoModel_16217796510031
// MI455X (gfx1250) — matrix operand order repaired and checked by reading, not compiled, not run on hardware
//
#include <hip/hip_runtime.h>
#include <hip/hip_bf16.h>
#include <stdint.h>

// CRF forward (partition - score) for B=512, S=512, T=78 on gfx1250.
//
// DP step in exp2/log2 domain (raw v_exp_f32 / v_log_f32 are base-2):
//   p = exp2(st2 - rowmax2);  q = p x exp(trans)  (WMMA f16->f32 GEMM)
//   st2' = em/ln2 + rowmax2 + log2(q)
// One workgroup = 16 batch rows, 2 waves: wave0 = alpha (unmasked), wave1 = score (masked).
// exp(trans) WMMA B-tiles live in LDS (shared by both waves) to stay under 256 VGPRs.
// Emissions double-buffered into LDS via the Tensor Data Mover (TENSORcnt-waited).

#define BATCH 512
#define SEQ   512
#define TAGS  78
#define NT    5    // 5 n-tiles * 16 = 80 output tags (padded)
#define KT    3    // 3 k-tiles * 32 = 96 contraction (padded)
#define PSTR  100  // halves per row in A-staging LDS (bank-conflict pad)

#define INV_LN2 1.44269504088896340736f
#define LN2     0.69314718055994530942f

typedef __attribute__((ext_vector_type(16))) _Float16     v16h;
typedef __attribute__((ext_vector_type(8)))  float        v8f;
typedef __attribute__((ext_vector_type(4)))  unsigned int v4u;
typedef __attribute__((ext_vector_type(2)))  unsigned int v2u;
typedef __attribute__((ext_vector_type(8)))  int          v8i;
typedef __attribute__((ext_vector_type(4)))  int          v4i;

#if __has_builtin(__builtin_amdgcn_tensor_load_to_lds) && __has_builtin(__builtin_amdgcn_s_wait_tensorcnt)
#define HAVE_TDM 1
#else
#define HAVE_TDM 0
#endif

#if __has_builtin(__builtin_amdgcn_exp2f)
#define EXP2F(x) __builtin_amdgcn_exp2f(x)   // raw v_exp_f32
#else
#define EXP2F(x) exp2f(x)
#endif
#if __has_builtin(__builtin_amdgcn_logf)
#define LOG2F(x) __builtin_amdgcn_logf(x)    // raw v_log_f32 (base-2)
#else
#define LOG2F(x) log2f(x)
#endif

// Load one step's emissions slab [16 rows][78 f32] (row stride SEQ*TAGS) into LDS.
__device__ __forceinline__ void load_em_slab(const float* __restrict__ em,
                                             int b0, int s, float* dst, int lane)
{
#if HAVE_TDM
  // Tensor DMA descriptor (D#), 2D tile: dim0 = 78 contiguous f32 (312B bursts), dim1 = 16 rows.
  unsigned long long ga = (unsigned long long)(uintptr_t)em
                        + 4ull * (((unsigned long long)b0 * SEQ + (unsigned)s) * TAGS);
  unsigned int lds = (unsigned int)(uintptr_t)dst;   // generic addr low 32 bits == LDS offset
  v4u g0;
  g0.x = 0x1u;                                       // count=1, user descriptor
  g0.y = lds;                                        // lds_addr (bytes)
  g0.z = (unsigned int)ga;                           // global_addr[31:0]
  g0.w = ((unsigned int)(ga >> 32) & 0x01FFFFFFu) | 0x80000000u;  // addr[56:32] | type=2
  v8i g1;
  g1[0] = 0x00020000;                                // data_size = 4B
  g1[1] = (TAGS & 0xFFFF) << 16;                     // tensor_dim0[15:0]
  g1[2] = (16 << 16);                                // tensor_dim0 hi = 0 | tensor_dim1 lo = 16
  g1[3] = (TAGS << 16);                              // tensor_dim1 hi = 0 | tile_dim0 = 78
  g1[4] = 16;                                        // tile_dim1 = 16, tile_dim2 = 0
  g1[5] = SEQ * TAGS;                                // tensor_dim0_stride = 39936 elements
  g1[6] = 0;
  g1[7] = 0;
  v4i z4 = {0, 0, 0, 0};
#if defined(__clang_major__) && (__clang_major__ >= 23)
  v8i z8 = {0, 0, 0, 0, 0, 0, 0, 0};
  __builtin_amdgcn_tensor_load_to_lds(g0, g1, z4, z4, z8, 0);
#else
  __builtin_amdgcn_tensor_load_to_lds(g0, g1, z4, z4, 0);
#endif
  (void)lane;
#else
  const float* src = em + ((size_t)b0 * SEQ + s) * TAGS;
  for (int i = lane; i < 16 * TAGS; i += 32) {
    int r = i / TAGS, c = i - r * TAGS;
    dst[i] = src[(size_t)r * SEQ * TAGS + c];
  }
#endif
}

__global__ __launch_bounds__(64) void crf_fwd_kernel(
    const float* __restrict__ em, const unsigned char* __restrict__ mask,
    const float* __restrict__ startT, const float* __restrict__ endT,
    const float* __restrict__ trans, float* __restrict__ out)
{
  __shared__ alignas(16) float    embuf[2][16 * TAGS];      // TDM double buffer
  __shared__ alignas(16) _Float16 pbuf[2][16 * PSTR];       // per-wave A-operand staging
  __shared__ alignas(32) _Float16 Blds[KT * NT * 32 * 16];  // exp(trans) B-tiles, shared
  __shared__ unsigned int         maskw[16 * SEQ / 4];      // 16 rows x 512 bool bytes
  __shared__ float                res[2][16];

  const int tid  = threadIdx.x;
  const int wave = tid >> 5;            // 0 = alpha, 1 = masked score
  const int lane = tid & 31;
  const int m    = lane & 15;           // column / row-in-tile index
  const int h    = lane >> 4;           // half-wave select
  const int b0   = blockIdx.x * 16;

  // ---- stage mask slab (contiguous 8 KB) ----
  {
    const unsigned int* g = (const unsigned int*)(mask + (size_t)b0 * SEQ);
    for (int i = tid; i < 16 * SEQ / 4; i += 64) maskw[i] = g[i];
  }
  // zero own A-staging once (covers K = 80..95 contraction padding forever)
  for (int i = lane; i < 16 * PSTR; i += 32) pbuf[wave][i] = (_Float16)0.f;

  // ---- B operand: exp(transitions) in WMMA B layout, built once in LDS ----
  // Tile (kt,nt), lane l holds column n = l&15; element e -> K = kt*32 + 16*(l>>4) + e.
  for (int idx = tid; idx < KT * NT * 32 * 16; idx += 64) {
    int e  = idx & 15;
    int l  = (idx >> 4) & 31;
    int t  = idx >> 9;                  // tile = kt*NT + nt
    int kt = t / NT, nt = t - kt * NT;
    int j  = kt * 32 + 16 * (l >> 4) + e;
    int k  = nt * 16 + (l & 15);
    float v = (j < TAGS && k < TAGS) ? EXP2F(trans[j * TAGS + k] * INV_LN2) : 0.f;
    Blds[idx] = (_Float16)v;
  }

  // ---- init state in log2 domain (C/D layout: row = v + 8*h, col = nt*16 + m) ----
  v8f st[NT];
  #pragma unroll
  for (int nt = 0; nt < NT; ++nt)
    #pragma unroll
    for (int v = 0; v < 8; ++v) {
      int row = v + 8 * h, k = nt * 16 + m;
      float x = -__builtin_inff();
      if (k < TAGS)
        x = (startT[k] + em[((size_t)(b0 + row) * SEQ) * TAGS + k]) * INV_LN2;
      st[nt][v] = x;
    }

  if (wave == 0) load_em_slab(em, b0, 1, &embuf[0][0], lane);   // prefetch step 1

  const v16h* Bl  = (const v16h*)Blds;
  const v2u*  P64 = (const v2u*)&pbuf[wave][0];

  for (int s = 1; s < SEQ; ++s) {
#if HAVE_TDM
    if (wave == 0) __builtin_amdgcn_s_wait_tensorcnt(0);
#endif
    __syncthreads();                                            // embuf[(s-1)&1] + Blds ready
    if (wave == 0 && s + 1 < SEQ)
      load_em_slab(em, b0, s + 1, &embuf[s & 1][0], lane);      // prefetch next (disjoint buf)
    const float* emb = &embuf[(s - 1) & 1][0];

    // branchless per-row update predicate (hoisted; wave0 always updates)
    bool upd[8];
    #pragma unroll
    for (int v = 0; v < 8; ++v) {
      int row = v + 8 * h;
      unsigned int mb = (maskw[(row * SEQ + s) >> 2] >> ((s & 3) * 8)) & 0xFFu;
      upd[v] = (wave == 0) | (mb != 0u);
    }

    // per-row max of state: 4 tile-maxes + 4 xor-shuffles inside each 16-lane half
    float rm[8];
    #pragma unroll
    for (int v = 0; v < 8; ++v) {
      float x = st[0][v];
      #pragma unroll
      for (int nt = 1; nt < NT; ++nt) x = fmaxf(x, st[nt][v]);
      x = fmaxf(x, __shfl_xor(x, 1, 32));
      x = fmaxf(x, __shfl_xor(x, 2, 32));
      x = fmaxf(x, __shfl_xor(x, 4, 32));
      x = fmaxf(x, __shfl_xor(x, 8, 32));
      rm[v] = x;
    }

    // p = exp2(st2 - rowmax2) -> f16 staging, [row][tag] row-major (transpose via LDS)
    _Float16* P = &pbuf[wave][0];
    #pragma unroll
    for (int nt = 0; nt < NT; ++nt)
      #pragma unroll
      for (int v = 0; v < 8; ++v)
        P[(v + 8 * h) * PSTR + nt * 16 + m] = (_Float16)EXP2F(st[nt][v] - rm[v]);

    // gather A tiles: lane m holds row m; 4-half groups are 8B aligned -> ds_load_b64
    // e in {4g..4g+3} -> K = 4g + 16h
    v16h A[KT];
    #pragma unroll
    for (int kt = 0; kt < KT; ++kt)
      #pragma unroll
      for (int g = 0; g < 4; ++g) {
        int Kl = 4 * g + 16 * h;
        v2u w = P64[(m * PSTR + kt * 32 + Kl) >> 2];
        union { v2u u; _Float16 f[4]; } cv; cv.u = w;
        #pragma unroll
        for (int q = 0; q < 4; ++q) A[kt][4 * g + q] = cv.f[q];
      }

    // q = P x expT : 5 n-tiles x 3 chained k-tiles of v_wmma_f32_16x16x32_f16
    #pragma unroll
    for (int nt = 0; nt < NT; ++nt) {
      v8f acc = {};
      #pragma unroll
      for (int kt = 0; kt < KT; ++kt) {
        v16h b = Bl[(kt * NT + nt) * 32 + lane];                // ds_load_b128 x2
        acc = __builtin_amdgcn_wmma_f32_16x16x32_f16(
                  false, A[kt], false, b, (short)0, acc, false, false);
      }
      #pragma unroll
      for (int v = 0; v < 8; ++v) {
        int row = v + 8 * h, col = nt * 16 + m;
        float ns = -__builtin_inff();
        if (col < TAGS)
          ns = fmaf(emb[row * TAGS + col], INV_LN2, rm[v] + LOG2F(acc[v]));
        st[nt][v] = upd[v] ? ns : st[nt][v];                    // v_cndmask, no exec churn
      }
    }
  }

  // ---- epilogue (base-2): lse2 per row, out = (alpha_lse2 - score_lse2) * ln2 ----
  #pragma unroll
  for (int v = 0; v < 8; ++v) {
    float x = -__builtin_inff();
    #pragma unroll
    for (int nt = 0; nt < NT; ++nt) {
      int col = nt * 16 + m;
      float y = (col < TAGS) ? fmaf(endT[col], INV_LN2, st[nt][v]) : -__builtin_inff();
      x = fmaxf(x, y);
    }
    x = fmaxf(x, __shfl_xor(x, 1, 32));
    x = fmaxf(x, __shfl_xor(x, 2, 32));
    x = fmaxf(x, __shfl_xor(x, 4, 32));
    x = fmaxf(x, __shfl_xor(x, 8, 32));
    float sacc = 0.f;
    #pragma unroll
    for (int nt = 0; nt < NT; ++nt) {
      int col = nt * 16 + m;
      if (col < TAGS) sacc += EXP2F(fmaf(endT[col], INV_LN2, st[nt][v]) - x);
    }
    sacc += __shfl_xor(sacc, 1, 32);
    sacc += __shfl_xor(sacc, 2, 32);
    sacc += __shfl_xor(sacc, 4, 32);
    sacc += __shfl_xor(sacc, 8, 32);
    if (m == 0) res[wave][v + 8 * h] = x + LOG2F(sacc);
  }
  __syncthreads();
  if (tid < 16) out[b0 + tid] = (res[0][tid] - res[1][tid]) * LN2;
}

extern "C" void kernel_launch(void* const* d_in, const int* in_sizes, int n_in,
                              void* d_out, int out_size, void* d_ws, size_t ws_size,
                              hipStream_t stream)
{
  const float*         em     = (const float*)d_in[0];
  const unsigned char* mask   = (const unsigned char*)d_in[1];   // bool array
  const float*         startT = (const float*)d_in[2];
  const float*         endT   = (const float*)d_in[3];
  const float*         trans  = (const float*)d_in[4];
  float*               out    = (float*)d_out;
  crf_fwd_kernel<<<dim3(BATCH / 16), dim3(64), 0, stream>>>(em, mask, startT, endT, trans, out);
  (void)in_sizes; (void)n_in; (void)out_size; (void)d_ws; (void)ws_size;
}
